// Universal_19799799234807
// MI455X (gfx1250) — matrix operand order repaired and checked by reading, not compiled, not run on hardware
//
#include <hip/hip_runtime.h>

#define N_NODES  100000
#define N_EDGES  1600000
#define FEATS    512
#define HIDDEN   64
#define CLASSES  16
#define DEPTH    10
#define DIFF_F   0.9f

typedef __attribute__((ext_vector_type(16))) __bf16 v16bf;
typedef __attribute__((ext_vector_type(8)))  float  v8f;

__device__ __forceinline__ v8f wmma_bf16(v16bf a, v16bf b, v8f c) {
  // 8 args: (neg_a, A, neg_b, B, c_mod, C, reuse_a, reuse_b)
  return __builtin_amdgcn_wmma_f32_16x16x32_bf16(false, a, false, b, (short)0, c,
                                                 false, false);
}

// ---------------------------------------------------------------- utility ---
__global__ void k_zero(float* __restrict__ p, int n) {
  int i = blockIdx.x * blockDim.x + threadIdx.x;
  if (i < n) p[i] = 0.0f;
}

__global__ void k_deg(const int* __restrict__ dst, float* __restrict__ deg, int e) {
  int i = blockIdx.x * blockDim.x + threadIdx.x;
  if (i < e) atomicAdd(&deg[dst[i]], 1.0f);
}

__global__ void k_dinv(float* __restrict__ deg, int n) {  // in-place deg -> dinv
  int i = blockIdx.x * blockDim.x + threadIdx.x;
  if (i < n) {
    float d = deg[i];
    deg[i] = (d > 0.0f) ? rsqrtf(fmaxf(d, 1.0f)) : 0.0f;
  }
}

__global__ void k_wd(const int* __restrict__ src, const int* __restrict__ dst,
                     const float* __restrict__ dinv, float* __restrict__ wd, int e) {
  int i = blockIdx.x * blockDim.x + threadIdx.x;
  if (i < e) wd[i] = dinv[src[i]] * dinv[dst[i]] * DIFF_F;  // fold DIFF into weight
}

// ------------------------------------------- pack W into WMMA B-reg layout ---
// For each (kblock, ntile, lane, j): element = W[k, col] with
//   k = kblock*32 + (lane>>4)*8 + (j&7) + (j>>3)*16,  col = ntile*16 + (lane&15)
// stored contiguously so a lane fetches its whole v16bf operand in one 32B load.
__global__ void k_pack(const float* __restrict__ W, __bf16* __restrict__ hi,
                       __bf16* __restrict__ lo, int K, int Ncols) {
  int idx = blockIdx.x * blockDim.x + threadIdx.x;
  if (idx >= K * Ncols) return;
  int j      = idx & 15;
  int lane   = (idx >> 4) & 31;
  int rest   = idx >> 9;
  int ntiles = Ncols >> 4;
  int nt     = rest % ntiles;
  int kb     = rest / ntiles;
  int k   = kb * 32 + (lane >> 4) * 8 + (j & 7) + (j >> 3) * 16;
  int col = nt * 16 + (lane & 15);
  float v = W[k * Ncols + col];
  __bf16 h = (__bf16)v;
  hi[idx] = h;
  lo[idx] = (__bf16)(v - (float)h);
}

// --------------------------------------------------- GEMM1: h = x@W_dim + b ---
// One wave computes a full 16x64 strip (4 accumulator tiles): A (x) is read
// from HBM exactly once; B comes pre-packed as hi/lo bf16 vector operands.
__global__ void __launch_bounds__(256)
k_gemm_dim(const float* __restrict__ A, const __bf16* __restrict__ Bhi,
           const __bf16* __restrict__ Blo, const float* __restrict__ bias,
           float* __restrict__ C) {
  const int wave  = threadIdx.x >> 5;
  const int lane  = threadIdx.x & 31;
  const int strip = blockIdx.x * 8 + wave;
  if (strip >= N_NODES / 16) return;   // wave-uniform: EXEC stays all-ones
  const int m0   = strip * 16;
  const int rowA = m0 + (lane & 15);
  const int kbA  = (lane >> 4) * 8;    // A layout: halves 0-7 K=kb.., 8-15 K=16+kb..

  v8f acc[4] = {{}, {}, {}, {}};
  const float* ap = A + rowA * FEATS + kbA;
  const v16bf* bh0 = (const v16bf*)Bhi + lane;
  const v16bf* bl0 = (const v16bf*)Blo + lane;

  for (int kb = 0; kb < FEATS / 32; ++kb) {
    const float4 a0 = *(const float4*)(ap + 0);
    const float4 a1 = *(const float4*)(ap + 4);
    const float4 a2 = *(const float4*)(ap + 16);
    const float4 a3 = *(const float4*)(ap + 20);
    const float av[16] = {a0.x, a0.y, a0.z, a0.w, a1.x, a1.y, a1.z, a1.w,
                          a2.x, a2.y, a2.z, a2.w, a3.x, a3.y, a3.z, a3.w};
    v16bf ahi, alo;
#pragma unroll
    for (int j = 0; j < 16; ++j) {
      __bf16 h = (__bf16)av[j];
      ahi[j] = h;
      alo[j] = (__bf16)(av[j] - (float)h);
    }
#pragma unroll
    for (int nt = 0; nt < 4; ++nt) {
      const v16bf bh = bh0[(kb * 4 + nt) * 32];
      const v16bf bl = bl0[(kb * 4 + nt) * 32];
      // split-precision: A*B ~= Ah*Bh + Al*Bh + Ah*Bl (f32 accumulate)
      acc[nt] = wmma_bf16(ahi, bh, acc[nt]);
      acc[nt] = wmma_bf16(alo, bh, acc[nt]);
      acc[nt] = wmma_bf16(ahi, bl, acc[nt]);
    }
    ap += 32;
  }

  const int mbase = m0 + (lane >> 4) * 8;  // C layout: lanes16-31 hold rows M+8
#pragma unroll
  for (int nt = 0; nt < 4; ++nt) {
    const int col = nt * 16 + (lane & 15);
    const float bv = bias[col];
#pragma unroll
    for (int r = 0; r < 8; ++r)
      C[(mbase + r) * HIDDEN + col] = acc[nt][r] + bv;
  }
}

// ------------------------------------------------ GEMM2: out = h@W_cls + b ---
__global__ void __launch_bounds__(256)
k_gemm_cls(const float* __restrict__ A, const __bf16* __restrict__ Bhi,
           const __bf16* __restrict__ Blo, const float* __restrict__ bias,
           float* __restrict__ C) {
  const int wave  = threadIdx.x >> 5;
  const int lane  = threadIdx.x & 31;
  const int strip = blockIdx.x * 8 + wave;
  if (strip >= N_NODES / 16) return;
  const int m0   = strip * 16;
  const int rowA = m0 + (lane & 15);
  const int kbA  = (lane >> 4) * 8;

  v8f acc = {};
  const float* ap = A + rowA * HIDDEN + kbA;
#pragma unroll
  for (int kb = 0; kb < HIDDEN / 32; ++kb) {
    const float4 a0 = *(const float4*)(ap + 0);
    const float4 a1 = *(const float4*)(ap + 4);
    const float4 a2 = *(const float4*)(ap + 16);
    const float4 a3 = *(const float4*)(ap + 20);
    const float av[16] = {a0.x, a0.y, a0.z, a0.w, a1.x, a1.y, a1.z, a1.w,
                          a2.x, a2.y, a2.z, a2.w, a3.x, a3.y, a3.z, a3.w};
    v16bf ahi, alo;
#pragma unroll
    for (int j = 0; j < 16; ++j) {
      __bf16 h = (__bf16)av[j];
      ahi[j] = h;
      alo[j] = (__bf16)(av[j] - (float)h);
    }
    const v16bf bh = *((const v16bf*)Bhi + kb * 32 + lane);
    const v16bf bl = *((const v16bf*)Blo + kb * 32 + lane);
    acc = wmma_bf16(ahi, bh, acc);
    acc = wmma_bf16(alo, bh, acc);
    acc = wmma_bf16(ahi, bl, acc);
    ap += 32;
  }

  const int col = lane & 15;
  const float bv = bias[col];
  const int mbase = m0 + (lane >> 4) * 8;
#pragma unroll
  for (int r = 0; r < 8; ++r)
    C[(mbase + r) * CLASSES + col] = acc[r] + bv;
}

// ------------------------------------------------------- diffusion kernels ---
__global__ void k_init_agg(const float* __restrict__ h0, float* __restrict__ agg, int n) {
  int i = blockIdx.x * blockDim.x + threadIdx.x;
  if (i < n) agg[i] = (1.0f - DIFF_F) * h0[i];
}

// 16 lanes per edge, 4 contiguous floats each: one float4 load + 4 L2 atomics.
// h (25.6 MB) is L2-resident; only the edge streams hit HBM -> prefetch them.
__global__ void __launch_bounds__(256)
k_scatter(const int* __restrict__ src, const int* __restrict__ dst,
          const float* __restrict__ wd, const float* __restrict__ h,
          float* __restrict__ agg) {
  int gid  = blockIdx.x * blockDim.x + threadIdx.x;
  int e    = gid >> 4;
  int part = gid & 15;
  if (e >= N_EDGES) return;
  if (part == 0) {
    int ep = (e + 4096 < N_EDGES) ? e + 4096 : e;
    __builtin_prefetch(src + ep, 0, 1);   // global_prefetch_b8
    __builtin_prefetch(dst + ep, 0, 1);
    __builtin_prefetch(wd + ep, 0, 1);
  }
  int   s = src[e];
  int   d = dst[e];
  float w = wd[e];
  const float4 hv = *(const float4*)(h + s * HIDDEN + part * 4);
  float* out = agg + d * HIDDEN + part * 4;
  atomicAdd(out + 0, w * hv.x);
  atomicAdd(out + 1, w * hv.y);
  atomicAdd(out + 2, w * hv.z);
  atomicAdd(out + 3, w * hv.w);
}

// ------------------------------------------------- per-element "attention" ---
// f = leaky_relu([h, h0, emb0] @ Wa1 + ba1) @ Wa2 + ba2, scaled by 0.5.
// emb0 rows are a broadcast constant -> pure elementwise per (node, feature).
__global__ void k_mlp(const float* __restrict__ h, const float* __restrict__ h0,
                      const float* __restrict__ Wa1, const float* __restrict__ ba1,
                      const float* __restrict__ Wa2, const float* __restrict__ ba2,
                      const float* __restrict__ emb, float* __restrict__ out, int n) {
  int i = blockIdx.x * blockDim.x + threadIdx.x;
  if (i >= n) return;
  float xv  = h[i];
  float h0v = h0[i];
  float acc = ba2[0];
#pragma unroll
  for (int j = 0; j < 11; ++j) {
    float t = ba1[j] + xv * Wa1[j] + h0v * Wa1[11 + j];
#pragma unroll
    for (int m = 0; m < 7; ++m) t += emb[m] * Wa1[(2 + m) * 11 + j];
    t = (t > 0.0f) ? t : 0.01f * t;       // leaky_relu, slope 0.01
    acc += t * Wa2[j];
  }
  out[i] = acc * 0.5f;
}

// --------------------------------------------------------------- dispatch ---
extern "C" void kernel_launch(void* const* d_in, const int* in_sizes, int n_in,
                              void* d_out, int out_size, void* d_ws, size_t ws_size,
                              hipStream_t stream) {
  const float* x     = (const float*)d_in[0];
  const int*   edges = (const int*)d_in[1];
  const float* W_dim = (const float*)d_in[2];
  const float* b_dim = (const float*)d_in[3];
  const float* emb   = (const float*)d_in[4];
  const float* W_a1  = (const float*)d_in[5];
  const float* b_a1  = (const float*)d_in[6];
  const float* W_a2  = (const float*)d_in[7];
  const float* b_a2  = (const float*)d_in[8];
  const float* W_cls = (const float*)d_in[9];
  const float* b_cls = (const float*)d_in[10];
  float*       out   = (float*)d_out;

  const int* src  = edges;             // edges[0, :]
  const int* dstE = edges + N_EDGES;   // edges[1, :]

  // ws layout (floats): dinv[N] | wd[E] | buf0 | buf1 | buf2 | packed weights
  float* dinv = (float*)d_ws;
  float* wd   = dinv + N_NODES;
  float* buf0 = wd + N_EDGES;
  float* buf1 = buf0 + (size_t)N_NODES * HIDDEN;
  float* buf2 = buf1 + (size_t)N_NODES * HIDDEN;
  __bf16* wdim_hi = (__bf16*)(buf2 + (size_t)N_NODES * HIDDEN);
  __bf16* wdim_lo = wdim_hi + FEATS * HIDDEN;
  __bf16* wcls_hi = wdim_lo + FEATS * HIDDEN;
  __bf16* wcls_lo = wcls_hi + HIDDEN * CLASSES;

  const int NH = N_NODES * HIDDEN;                 // 6.4M
  const int nb_N  = (N_NODES + 255) / 256;
  const int nb_E  = (N_EDGES + 255) / 256;
  const int nb_NH = (NH + 255) / 256;
  const int nb_SC = (N_EDGES * 16) / 256;          // exact: E*16 % 256 == 0
  const int nb_ST = (N_NODES / 16 + 7) / 8;        // 6250 strips, 8 waves/block

  // degrees -> dinv -> cached edge weights (DIFF folded in)
  k_zero<<<nb_N, 256, 0, stream>>>(dinv, N_NODES);
  k_deg <<<nb_E, 256, 0, stream>>>(dstE, dinv, N_EDGES);
  k_dinv<<<nb_N, 256, 0, stream>>>(dinv, N_NODES);
  k_wd  <<<nb_E, 256, 0, stream>>>(src, dstE, dinv, wd, N_EDGES);

  // pre-pack weights into WMMA B-register layout (hi/lo bf16 split)
  k_pack<<<(FEATS * HIDDEN + 255) / 256, 256, 0, stream>>>(W_dim, wdim_hi, wdim_lo,
                                                           FEATS, HIDDEN);
  k_pack<<<(HIDDEN * CLASSES + 255) / 256, 256, 0, stream>>>(W_cls, wcls_hi, wcls_lo,
                                                             HIDDEN, CLASSES);

  // h0 = x @ W_dim + b_dim
  k_gemm_dim<<<nb_ST, 256, 0, stream>>>(x, wdim_hi, wdim_lo, b_dim, buf0);

  // diffusion phase 1: h = 0.9*conv(h) + 0.1*h0
  const float* cur = buf0;
  for (int it = 0; it < DEPTH; ++it) {
    float* nxt = (it & 1) ? buf2 : buf1;
    k_init_agg<<<nb_NH, 256, 0, stream>>>(buf0, nxt, NH);
    k_scatter <<<nb_SC, 256, 0, stream>>>(src, dstE, wd, cur, nxt);
    cur = nxt;
  }

  // elementwise MLP (writes phase-2 h0 in-place over buf0)
  k_mlp<<<nb_NH, 256, 0, stream>>>(cur, buf0, W_a1, b_a1, W_a2, b_a2, emb, buf0, NH);

  // diffusion phase 2
  cur = buf0;
  for (int it = 0; it < DEPTH; ++it) {
    float* nxt = (it & 1) ? buf2 : buf1;
    k_init_agg<<<nb_NH, 256, 0, stream>>>(buf0, nxt, NH);
    k_scatter <<<nb_SC, 256, 0, stream>>>(src, dstE, wd, cur, nxt);
    cur = nxt;
  }

  // out = h @ W_cls + b_cls
  k_gemm_cls<<<nb_ST, 256, 0, stream>>>(cur, wcls_hi, wcls_lo, b_cls, out);
}
